// Arch13GraphEncoder_72713796321738
// MI455X (gfx1250) — compile-verified, matrix operand order checked by reading, not run on hardware
//
#include <hip/hip_runtime.h>
#include <hip/hip_bf16.h>
#include <math.h>

// ---------------- problem constants (match reference) ----------------
constexpr int N_  = 100000;   // nodes
constexpr int E_  = 200000;   // edges
constexpr int H_  = 128;      // hidden
constexpr int L_  = 6;        // GIN layers
constexpr int T_  = 4;        // transformer layers
constexpr int S_  = 4096;     // B*M tokens
constexpr int B_  = 64;
constexpr int M_  = 64;
constexpr int K_  = 16;       // graphlet size
constexpr int NH_ = 4;
constexpr int HD_ = 32;       // head dim
constexpr int H2_ = 256;      // ffn hidden
constexpr int H3_ = 384;      // qkv width

typedef __bf16 bf16_t;
typedef __attribute__((ext_vector_type(16))) __bf16 v16bf;
typedef __attribute__((ext_vector_type(8)))  __bf16 v8bf;
typedef __attribute__((ext_vector_type(8)))  float  v8f;
typedef unsigned int u32x4 __attribute__((ext_vector_type(4)));
typedef int          i32x8 __attribute__((ext_vector_type(8)));
typedef int          i32x4 __attribute__((ext_vector_type(4)));

// ---------------- workspace layout ----------------
constexpr size_t AL(size_t x) { return (x + 255) & ~size_t(255); }
constexpr size_t o_h     = 0;                                        // f32 [N,H]
constexpr size_t o_agg   = AL(o_h     + (size_t)N_*H_*4);            // f32 [N,H] (agg, then t2)
constexpr size_t o_xb    = AL(o_agg   + (size_t)N_*H_*4);            // bf16 [N,H]
constexpr size_t o_t1b   = AL(o_xb    + (size_t)N_*H_*2);            // bf16 [N,H]
constexpr size_t o_stats = AL(o_t1b   + (size_t)N_*H_*2);            // f32 [2,H]
constexpr size_t o_wg1   = AL(o_stats + (size_t)2*H_*4);             // bf16 [L,H,H]  (transposed)
constexpr size_t o_wg2   = AL(o_wg1   + (size_t)L_*H_*H_*2);
constexpr size_t o_tokt  = AL(o_wg2   + (size_t)L_*H_*H_*2);         // bf16 [H,H]
constexpr size_t o_lpt   = AL(o_tokt  + (size_t)H_*H_*2);            // bf16 [H,H]
constexpr size_t o_qkvt  = AL(o_lpt   + (size_t)H_*H_*2);            // bf16 [T,H3,H]
constexpr size_t o_outt  = AL(o_qkvt  + (size_t)T_*H_*H3_*2);        // bf16 [T,H,H]
constexpr size_t o_f1t   = AL(o_outt  + (size_t)T_*H_*H_*2);         // bf16 [T,H2,H]
constexpr size_t o_f2t   = AL(o_f1t   + (size_t)T_*H2_*H_*2);        // bf16 [T,H,H2]
constexpr size_t o_zb    = AL(o_f2t   + (size_t)T_*H_*H2_*2);        // bf16 [S,H]
constexpr size_t o_hidb  = AL(o_zb    + (size_t)S_*H_*2);            // bf16 [S,H]
constexpr size_t o_ht    = AL(o_hidb  + (size_t)S_*H_*2);            // f32 [S,H]
constexpr size_t o_rb    = AL(o_ht    + (size_t)S_*H_*4);            // bf16 [S,H]
constexpr size_t o_qkvb  = AL(o_rb    + (size_t)S_*H_*2);            // f32 [S,H3]
constexpr size_t o_aob   = AL(o_qkvb  + (size_t)S_*H3_*4);           // bf16 [S,H]
constexpr size_t o_f1b   = AL(o_aob   + (size_t)S_*H_*2);            // bf16 [S,H2]

// ====================================================================
// Strip WMMA GEMM: one wave computes a 16-row x 128-col strip of
// out[M,Nc] = act(X[M,KD] @ Wt[Nc,KD]^T + bias) (+accum).
// - A fragment loaded once per k-step, reused across 8 column tiles.
// - B strip (128 x 128 k-chunk of pre-transposed weights) staged into
//   LDS by the Tensor Data Mover (tensor_load_to_lds + s_wait_tensorcnt),
//   with TDM row padding (16B per 256B row) to rotate LDS banks.
// - A fragment (ISA 7.12.2, 16-bit A 16x32): lanes 0-15 row=lane,
//   K=k0..k0+7 & k0+16..k0+23; lanes 16-31 row=lane-16, +8 on K.
// ====================================================================
constexpr int RLEN  = 128;   // k-chunk staged in LDS
constexpr int PITCH = 136;   // padded LDS row pitch (elements): 256B + 16B pad

template<int KD>
__global__ __launch_bounds__(32)
void wmma_gemm_strip(const bf16_t* __restrict__ X,
                     const bf16_t* __restrict__ Wt,
                     const float*  __restrict__ bias,
                     float*        __restrict__ outF,
                     bf16_t*       __restrict__ outB,
                     float*        __restrict__ stats,
                     int Ncol, int act, int accum)
{
  __shared__ alignas(16) bf16_t lb[128 * PITCH];
  const int lane  = threadIdx.x & 31;
  const int sub   = lane & 15;
  const int half  = lane >> 4;
  const int koff  = half * 8;
  const int arow  = blockIdx.x * 16 + sub;
  const int nbase = blockIdx.y * 128;

  const v8f vzero = {0.f,0.f,0.f,0.f,0.f,0.f,0.f,0.f};
  v8f acc[8];
#pragma unroll
  for (int n = 0; n < 8; ++n) acc[n] = vzero;

#pragma unroll
  for (int kc = 0; kc < KD; kc += RLEN) {
    __syncthreads();   // protect LDS reuse across k-chunks
    const bf16_t* gB = Wt + (size_t)nbase * KD + kc;
#if __has_builtin(__builtin_amdgcn_tensor_load_to_lds)
    {
      // D# group0: count=1 | lds_addr (array at LDS offset 0) | global addr | type=2
      const unsigned long long ga = (unsigned long long)(uintptr_t)gB;
      u32x4 g0 = { 1u, 0u, (unsigned)ga,
                   (unsigned)((ga >> 32) & 0x01FFFFFFu) | 0x80000000u };
      // D# group1: data_size=2B, pad_enable, pad_interval=64DW, pad_amount=4DW;
      // tensor_dim0=RLEN, tensor_dim1=128, tile=128xRLEN, dim0_stride=KD.
      i32x8 g1 = { (int)0x07510000u,
                   (int)((unsigned)RLEN << 16),   // tensor_dim0[15:0] @ [31:16]
                   (int)(128u << 16),             // tensor_dim1[15:0] @ [31:16]
                   (int)((unsigned)RLEN << 16),   // tile_dim0 @ [31:16]
                   128,                           // tile_dim1
                   KD,                            // tensor_dim0_stride[31:0]
                   0, 0 };
      i32x4 gz = {0, 0, 0, 0};
#if __clang_major__ >= 23
      i32x8 gz8 = {0, 0, 0, 0, 0, 0, 0, 0};
      __builtin_amdgcn_tensor_load_to_lds(g0, g1, gz, gz, gz8, 0);
#else
      __builtin_amdgcn_tensor_load_to_lds(g0, g1, gz, gz, 0);
#endif
      __builtin_amdgcn_s_wait_tensorcnt(0);
    }
#else
    // fallback: wave copies strip into LDS with the same padded pitch
    for (int i = lane; i < 128 * (RLEN / 8); i += 32) {
      const int row = i / (RLEN / 8);
      const int kk  = (i % (RLEN / 8)) * 8;
      *(v8bf*)&lb[row * PITCH + kk] = *(const v8bf*)(gB + (size_t)row * KD + kk);
    }
#endif
    __syncthreads();

    const bf16_t* ap = X + (size_t)arow * KD + kc + koff;
#pragma unroll
    for (int k0 = 0; k0 < RLEN; k0 += 32) {
      v8bf alo = *(const v8bf*)(ap + k0);
      v8bf ahi = *(const v8bf*)(ap + k0 + 16);
      v16bf a;
#pragma unroll
      for (int i = 0; i < 8; ++i) { a[i] = alo[i]; a[8 + i] = ahi[i]; }
#pragma unroll
      for (int n = 0; n < 8; ++n) {
        const bf16_t* bq = lb + (size_t)(n * 16 + sub) * PITCH + k0 + koff;
        v8bf blo = *(const v8bf*)bq;
        v8bf bhi = *(const v8bf*)(bq + 16);
        v16bf b;
#pragma unroll
        for (int i = 0; i < 8; ++i) { b[i] = blo[i]; b[8 + i] = bhi[i]; }
        acc[n] = __builtin_amdgcn_wmma_f32_16x16x32_bf16(
            false, a, false, b, (short)0, acc[n], false, false);
      }
    }
  }

  // D layout: lane holds col=sub (within tile), rows (half*8 + r)
  const int rowbase = blockIdx.x * 16 + half * 8;
#pragma unroll
  for (int n = 0; n < 8; ++n) {
    const int col = nbase + n * 16 + sub;
    const float bv = bias ? bias[col] : 0.0f;
    float ssum = 0.0f, ssq = 0.0f;
#pragma unroll
    for (int r = 0; r < 8; ++r) {
      const size_t idx = (size_t)(rowbase + r) * Ncol + col;
      float v = acc[n][r] + bv;
      if (act == 1)      v = fmaxf(v, 0.0f);
      else if (act == 2) v = 0.5f * v * (1.0f + erff(v * 0.70710678118654752f));
      if (accum && outF) v += outF[idx];
      if (outF) outF[idx] = v;
      if (outB) outB[idx] = (bf16_t)v;
      ssum += v; ssq += v * v;
    }
    if (stats) {
      atomicAdd(&stats[col], ssum);
      atomicAdd(&stats[Ncol + col], ssq);
    }
  }
}

// ---------------- elementwise / prep kernels ----------------
__global__ void conv_w_t(const float* __restrict__ W, bf16_t* __restrict__ Wt,
                         int Kd, int Ncol) {
  const size_t base = (size_t)blockIdx.z * Kd * Ncol;
  const int idx = blockIdx.x * blockDim.x + threadIdx.x;
  if (idx < Kd * Ncol) {
    const int n = idx / Kd, k = idx % Kd;     // Wt[n*Kd+k] = W[k*Ncol+n]
    Wt[base + idx] = (bf16_t)W[base + (size_t)k * Ncol + n];
  }
}

__global__ void zero_f32(float* __restrict__ p, long long n) {
  const long long i = (long long)blockIdx.x * blockDim.x + threadIdx.x;
  if (i < n) p[i] = 0.0f;
}

__global__ void embed_nodes(const int* __restrict__ x, const float* __restrict__ emb,
                            float* __restrict__ h) {
  const long long i = (long long)blockIdx.x * blockDim.x + threadIdx.x;
  if (i < (long long)N_ * H_) {
    const int n = (int)(i >> 7), c = (int)(i & 127);
    h[i] = emb[(size_t)x[n] * H_ + c];
  }
}

__global__ void gine_scatter(const float* __restrict__ h, const int* __restrict__ src,
                             const int* __restrict__ dst, const int* __restrict__ eattr,
                             const float* __restrict__ bond, float* __restrict__ agg) {
  const long long i = (long long)blockIdx.x * blockDim.x + threadIdx.x;
  if (i < (long long)E_ * H_) {
    const int e = (int)(i >> 7), c = (int)(i & 127);
    float m = h[(size_t)src[e] * H_ + c] + bond[(size_t)(eattr[e] - 1) * H_ + c];
    m = fmaxf(m, 0.0f);
    atomicAdd(&agg[(size_t)dst[e] * H_ + c], m);
  }
}

__global__ void fuse_xb(const float* __restrict__ h, const float* __restrict__ agg,
                        bf16_t* __restrict__ xb, float* __restrict__ stats) {
  if (blockIdx.x == 0 && threadIdx.x < 2 * H_) stats[threadIdx.x] = 0.0f;
  const long long i = (long long)blockIdx.x * blockDim.x + threadIdx.x;
  if (i < (long long)N_ * H_) xb[i] = (bf16_t)(h[i] + agg[i]);
}

__global__ void bn_apply(float* __restrict__ h, const float* __restrict__ t2,
                         const float* __restrict__ stats,
                         const float* __restrict__ g, const float* __restrict__ b) {
  const long long i = (long long)blockIdx.x * blockDim.x + threadIdx.x;
  if (i < (long long)N_ * H_) {
    const int c = (int)(i & 127);
    const float invN = 1.0f / (float)N_;
    const float mu  = stats[c] * invN;
    const float var = stats[H_ + c] * invN - mu * mu;
    h[i] += (t2[i] - mu) * rsqrtf(var + 1e-5f) * g[c] + b[c];
  }
}

__global__ void pool_kernel(const float* __restrict__ h, const int* __restrict__ ns,
                            bf16_t* __restrict__ zb) {
  const int s = blockIdx.x, c = threadIdx.x;   // 128 threads
  float acc = 0.0f; int cnt = 0;
#pragma unroll
  for (int k = 0; k < K_; ++k) {
    const int v = ns[s * K_ + k];
    if (v >= 0) { acc += h[(size_t)v * H_ + c]; cnt++; }
  }
  const float d = cnt > 0 ? (float)cnt : 1.0f;
  zb[(size_t)s * H_ + c] = (bf16_t)(acc / d);
}

__global__ void lpe_hidden(const float* __restrict__ lp, const float* __restrict__ w1,
                           const float* __restrict__ b1, bf16_t* __restrict__ out) {
  const long long i = (long long)blockIdx.x * blockDim.x + threadIdx.x;
  if (i < (long long)S_ * H_) {
    const int s = (int)(i >> 7), j = (int)(i & 127);
    const float l = fminf(fmaxf(lp[s], -30.0f), 0.0f);
    out[i] = (bf16_t)fmaxf(l * w1[j] + b1[j], 0.0f);
  }
}

__global__ void layernorm_kernel(const float* __restrict__ x, const float* __restrict__ g,
                                 const float* __restrict__ b, bf16_t* __restrict__ out) {
  const int s = blockIdx.x, c = threadIdx.x;   // 128 threads
  __shared__ float red[H_];
  const float v = x[(size_t)s * H_ + c];
  red[c] = v;
  __syncthreads();
  for (int o = 64; o > 0; o >>= 1) { if (c < o) red[c] += red[c + o]; __syncthreads(); }
  const float mu = red[0] * (1.0f / H_);
  __syncthreads();
  const float d = v - mu;
  red[c] = d * d;
  __syncthreads();
  for (int o = 64; o > 0; o >>= 1) { if (c < o) red[c] += red[c + o]; __syncthreads(); }
  const float var = red[0] * (1.0f / H_);
  out[(size_t)s * H_ + c] = (bf16_t)(d * rsqrtf(var + 1e-5f) * g[c] + b[c]);
}

__global__ __launch_bounds__(256)
void attention_kernel(const float* __restrict__ qkv, bf16_t* __restrict__ outb) {
  const int b = blockIdx.x / NH_, hd = blockIdx.x % NH_;
  __shared__ float q[M_][HD_], k[M_][HD_], v[M_][HD_];
  __shared__ float sc[M_][M_ + 1];
  const int tid = threadIdx.x;   // 256
  for (int i = tid; i < M_ * HD_; i += 256) {
    const int m = i >> 5, d = i & 31;
    const float* row = qkv + (size_t)(b * M_ + m) * H3_;
    q[m][d] = row[hd * HD_ + d];
    k[m][d] = row[H_ + hd * HD_ + d];
    v[m][d] = row[2 * H_ + hd * HD_ + d];
  }
  __syncthreads();
  const float scale = 0.17677669529663687f;   // 1/sqrt(32)
  for (int i = tid; i < M_ * M_; i += 256) {
    const int m = i >> 6, n = i & 63;
    float a = 0.0f;
#pragma unroll
    for (int d = 0; d < HD_; ++d) a += q[m][d] * k[n][d];
    sc[m][n] = a * scale;
  }
  __syncthreads();
  if (tid < M_) {
    const int m = tid;
    float mx = -1e30f;
    for (int n = 0; n < M_; ++n) mx = fmaxf(mx, sc[m][n]);
    float sm = 0.0f;
    for (int n = 0; n < M_; ++n) { const float e = __expf(sc[m][n] - mx); sc[m][n] = e; sm += e; }
    const float inv = 1.0f / sm;
    for (int n = 0; n < M_; ++n) sc[m][n] *= inv;
  }
  __syncthreads();
  for (int i = tid; i < M_ * HD_; i += 256) {
    const int m = i >> 5, d = i & 31;
    float o = 0.0f;
    for (int n = 0; n < M_; ++n) o += sc[m][n] * v[n][d];
    outb[(size_t)(b * M_ + m) * H_ + hd * HD_ + d] = (bf16_t)o;
  }
}

__global__ void readout_kernel(const float* __restrict__ ht, const float* __restrict__ lp,
                               const float* __restrict__ w1, const float* __restrict__ b1,
                               const float* __restrict__ w2, const float* __restrict__ b2,
                               float* __restrict__ out) {
  const int b = blockIdx.x, tid = threadIdx.x;   // 128 threads
  __shared__ float w[M_];
  if (tid < M_) {
    const int m = tid;
    float l = lp[b * M_ + m];
    l = fminf(fmaxf(l, -30.0f), 0.0f);
    float acc = b2[0];
    for (int j = 0; j < H_ / 2; ++j) acc += fmaxf(l * w1[j] + b1[j], 0.0f) * w2[j];
    w[m] = acc;
  }
  __syncthreads();
  if (tid == 0) {
    float mx = -1e30f;
    for (int m = 0; m < M_; ++m) mx = fmaxf(mx, w[m]);
    float sm = 0.0f;
    for (int m = 0; m < M_; ++m) { w[m] = __expf(w[m] - mx); sm += w[m]; }
    const float inv = 1.0f / sm;
    for (int m = 0; m < M_; ++m) w[m] *= inv;
  }
  __syncthreads();
  float acc = 0.0f;
  for (int m = 0; m < M_; ++m) acc += w[m] * ht[(size_t)(b * M_ + m) * H_ + tid];
  out[(size_t)b * H_ + tid] = acc;
}

// ====================================================================
extern "C" void kernel_launch(void* const* d_in, const int* in_sizes, int n_in,
                              void* d_out, int out_size, void* d_ws, size_t ws_size,
                              hipStream_t stream) {
  (void)in_sizes; (void)n_in; (void)out_size; (void)ws_size;
  const int*   x        = (const int*)d_in[0];
  const int*   ei       = (const int*)d_in[1];
  const int*   eattr    = (const int*)d_in[2];
  const int*   ns       = (const int*)d_in[3];
  /* d_in[4] batch unused */
  const float* log_probs= (const float*)d_in[5];
  const float* atom_emb = (const float*)d_in[6];
  const float* bond_emb = (const float*)d_in[7];
  const float* gin_W1   = (const float*)d_in[8];
  const float* gin_b1   = (const float*)d_in[9];
  const float* gin_W2   = (const float*)d_in[10];
  const float* gin_b2   = (const float*)d_in[11];
  const float* gin_gamma= (const float*)d_in[12];
  const float* gin_beta = (const float*)d_in[13];
  const float* tok_W    = (const float*)d_in[14];
  const float* tok_b    = (const float*)d_in[15];
  const float* lp_W1    = (const float*)d_in[16];
  const float* lp_b1    = (const float*)d_in[17];
  const float* lp_W2    = (const float*)d_in[18];
  const float* lp_b2    = (const float*)d_in[19];
  const float* ln1_g    = (const float*)d_in[20];
  const float* ln1_b    = (const float*)d_in[21];
  const float* qkv_W    = (const float*)d_in[22];
  const float* out_W    = (const float*)d_in[23];
  const float* out_b    = (const float*)d_in[24];
  const float* ln2_g    = (const float*)d_in[25];
  const float* ln2_b    = (const float*)d_in[26];
  const float* ffn_W1   = (const float*)d_in[27];
  const float* ffn_b1   = (const float*)d_in[28];
  const float* ffn_W2   = (const float*)d_in[29];
  const float* ffn_b2   = (const float*)d_in[30];
  const float* rd_W1    = (const float*)d_in[31];
  const float* rd_b1    = (const float*)d_in[32];
  const float* rd_W2    = (const float*)d_in[33];
  const float* rd_b2    = (const float*)d_in[34];
  float* out = (float*)d_out;

  char* ws = (char*)d_ws;
  float*  h     = (float*)(ws + o_h);
  float*  agg   = (float*)(ws + o_agg);      // reused as t2
  bf16_t* xb    = (bf16_t*)(ws + o_xb);
  bf16_t* t1b   = (bf16_t*)(ws + o_t1b);
  float*  stats = (float*)(ws + o_stats);
  bf16_t* wg1t  = (bf16_t*)(ws + o_wg1);
  bf16_t* wg2t  = (bf16_t*)(ws + o_wg2);
  bf16_t* tokt  = (bf16_t*)(ws + o_tokt);
  bf16_t* lpt   = (bf16_t*)(ws + o_lpt);
  bf16_t* qkvt  = (bf16_t*)(ws + o_qkvt);
  bf16_t* outt  = (bf16_t*)(ws + o_outt);
  bf16_t* f1t   = (bf16_t*)(ws + o_f1t);
  bf16_t* f2t   = (bf16_t*)(ws + o_f2t);
  bf16_t* zb    = (bf16_t*)(ws + o_zb);
  bf16_t* hidb  = (bf16_t*)(ws + o_hidb);
  float*  ht    = (float*)(ws + o_ht);
  bf16_t* rb    = (bf16_t*)(ws + o_rb);
  float*  qkvb  = (float*)(ws + o_qkvb);
  bf16_t* aob   = (bf16_t*)(ws + o_aob);
  bf16_t* f1b   = (bf16_t*)(ws + o_f1b);

  const int* src = ei;
  const int* dst = ei + E_;

  // ---- weight prep: transposed bf16 copies ----
  auto cw = [&](const float* W, bf16_t* Wt, int Kd, int Nc, int batch) {
    dim3 g((Kd * Nc + 255) / 256, 1, batch);
    conv_w_t<<<g, 256, 0, stream>>>(W, Wt, Kd, Nc);
  };
  cw(gin_W1, wg1t, H_, H_, L_);
  cw(gin_W2, wg2t, H_, H_, L_);
  cw(tok_W,  tokt, H_, H_, 1);
  cw(lp_W2,  lpt,  H_, H_, 1);
  cw(qkv_W,  qkvt, H_, H3_, T_);
  cw(out_W,  outt, H_, H_, T_);
  cw(ffn_W1, f1t,  H_, H2_, T_);
  cw(ffn_W2, f2t,  H2_, H_, T_);

  auto gemm = [&](const bf16_t* X, const bf16_t* Wt, const float* bias,
                  float* oF, bf16_t* oB, float* st,
                  int Mr, int Kd, int Nc, int act, int accum) {
    dim3 g(Mr / 16, Nc / 128);
    if (Kd == 128)
      wmma_gemm_strip<128><<<g, 32, 0, stream>>>(X, Wt, bias, oF, oB, st, Nc, act, accum);
    else
      wmma_gemm_strip<256><<<g, 32, 0, stream>>>(X, Wt, bias, oF, oB, st, Nc, act, accum);
  };

  const long long NH = (long long)N_ * H_;
  const long long EH = (long long)E_ * H_;

  // ---- node embedding ----
  embed_nodes<<<(int)((NH + 255) / 256), 256, 0, stream>>>(x, atom_emb, h);

  // ---- GINE layers ----
  for (int l = 0; l < L_; ++l) {
    zero_f32<<<(int)((NH + 255) / 256), 256, 0, stream>>>(agg, NH);
    gine_scatter<<<(int)((EH + 255) / 256), 256, 0, stream>>>(h, src, dst, eattr, bond_emb, agg);
    fuse_xb<<<(int)((NH + 255) / 256), 256, 0, stream>>>(h, agg, xb, stats);
    // t1 = relu(xb @ W1 + b1) -> bf16
    gemm(xb, wg1t + (size_t)l * H_ * H_, gin_b1 + l * H_,
         nullptr, t1b, nullptr, N_, H_, H_, /*relu*/1, 0);
    // t2 = relu(t1 @ W2 + b2) -> f32 (into agg buffer) + BN stats
    gemm(t1b, wg2t + (size_t)l * H_ * H_, gin_b2 + l * H_,
         agg, nullptr, stats, N_, H_, H_, /*relu*/1, 0);
    bn_apply<<<(int)((NH + 255) / 256), 256, 0, stream>>>(h, agg, stats,
         gin_gamma + l * H_, gin_beta + l * H_);
  }

  // ---- graphlet pooling + log-prob encoder ----
  pool_kernel<<<S_, H_, 0, stream>>>(h, ns, zb);
  lpe_hidden<<<(S_ * H_ + 255) / 256, 256, 0, stream>>>(log_probs, lp_W1, lp_b1, hidb);
  gemm(zb,   tokt, tok_b, ht, nullptr, nullptr, S_, H_, H_, 0, 0);
  gemm(hidb, lpt,  lp_b2, ht, nullptr, nullptr, S_, H_, H_, 0, 1);

  // ---- transformer layers ----
  for (int l = 0; l < T_; ++l) {
    layernorm_kernel<<<S_, H_, 0, stream>>>(ht, ln1_g + l * H_, ln1_b + l * H_, rb);
    gemm(rb, qkvt + (size_t)l * H_ * H3_, nullptr, qkvb, nullptr, nullptr,
         S_, H_, H3_, 0, 0);
    attention_kernel<<<B_ * NH_, 256, 0, stream>>>(qkvb, aob);
    gemm(aob, outt + (size_t)l * H_ * H_, out_b + l * H_, ht, nullptr, nullptr,
         S_, H_, H_, 0, /*accum*/1);
    layernorm_kernel<<<S_, H_, 0, stream>>>(ht, ln2_g + l * H_, ln2_b + l * H_, rb);
    gemm(rb, f1t + (size_t)l * H2_ * H_, ffn_b1 + l * H2_, nullptr, f1b, nullptr,
         S_, H_, H2_, /*gelu*/2, 0);
    gemm(f1b, f2t + (size_t)l * H_ * H2_, ffn_b2 + l * H_, ht, nullptr, nullptr,
         S_, H2_, H_, 0, /*accum*/1);
  }

  // ---- softmax-weighted readout ----
  readout_kernel<<<B_, H_, 0, stream>>>(ht, log_probs, rd_W1, rd_b1, rd_W2, rd_b2, out);
}
